// MatrixMultApproximator_73813307949484
// MI455X (gfx1250) — compile-verified, matrix operand order checked by reading, not run on hardware
//
#include <hip/hip_runtime.h>

typedef __attribute__((ext_vector_type(16))) _Float16 v16h;
typedef __attribute__((ext_vector_type(8)))  _Float16 v8h;
typedef __attribute__((ext_vector_type(4)))  _Float16 v4h;
typedef __attribute__((ext_vector_type(8)))  float    v8f;

#define O_DIM 4096
#define I_DIM 4096
#define B_DIM 4096

#define BM 128       // rows of X per block
#define BN 64        // weight rows (output cols) per block
#define BK 32        // K per WMMA step (fallback kernel)
#define LDSK 40      // padded LDS K-stride (halves) for fallback

#define BK2 64       // K per LDS stage (async kernel) = 2 WMMA substeps
#define LDSK2 72     // padded K-stride (halves): 144B rows, conflict-free
#define X_HALVES (BM * LDSK2)
#define W_HALVES (BN * LDSK2)
#define WP_OFF   (X_HALVES)
#define WT_OFF   (X_HALVES + W_HALVES)
#define STAGE_H  (X_HALVES + 2 * W_HALVES)

#define RED_BLOCKS 2048
#define RED_THREADS 256

// ------------------------- CDNA5 async helpers -------------------------------
__device__ __forceinline__ void wait_async0() {
#if __has_builtin(__builtin_amdgcn_s_wait_asynccnt)
  __builtin_amdgcn_s_wait_asynccnt(0);
#else
  asm volatile("s_wait_asynccnt 0x0" ::: "memory");
#endif
}

// GVS-mode async copy: LDS[vdst] <- MEM[saddr + voffset], 16 bytes, ASYNCcnt.
__device__ __forceinline__ void async_b128(unsigned int lds_addr,
                                           const _Float16* sbase,
                                           unsigned int byte_off) {
  asm volatile("global_load_async_to_lds_b128 %0, %1, %2"
               :
               : "v"(lds_addr), "v"(byte_off), "s"(sbase)
               : "memory");
}

// ---------------- Pass 1: partial sums of |V - Vp*cos(theta)| ----------------
__global__ __launch_bounds__(RED_THREADS)
void absmean_partial(const float* __restrict__ V, const float* __restrict__ Vp,
                     const float* __restrict__ theta_p, float* __restrict__ partials) {
  const float cos_t = cosf(theta_p[0]);
  __shared__ float red[RED_THREADS];
  float acc = 0.f;
  const size_t n = (size_t)O_DIM * I_DIM;
  for (size_t i = (size_t)blockIdx.x * RED_THREADS + threadIdx.x; i < n;
       i += (size_t)RED_BLOCKS * RED_THREADS) {
    acc += fabsf(V[i] - Vp[i] * cos_t);
  }
  red[threadIdx.x] = acc;
  __syncthreads();
  for (int s = RED_THREADS / 2; s > 0; s >>= 1) {
    if ((int)threadIdx.x < s) red[threadIdx.x] += red[threadIdx.x + s];
    __syncthreads();
  }
  if (threadIdx.x == 0) partials[blockIdx.x] = red[0];
}

// ---------------- Pass 2: finalize scale = 1/max(mean|T|, eps) ---------------
__global__ __launch_bounds__(RED_THREADS)
void finalize_scale(const float* __restrict__ partials, const float* __restrict__ theta_p,
                    float* __restrict__ scale_out) {
  __shared__ float red[RED_THREADS];
  float acc = 0.f;
  for (int i = threadIdx.x; i < RED_BLOCKS; i += RED_THREADS) acc += partials[i];
  red[threadIdx.x] = acc;
  __syncthreads();
  for (int s = RED_THREADS / 2; s > 0; s >>= 1) {
    if ((int)threadIdx.x < s) red[threadIdx.x] += red[threadIdx.x + s];
    __syncthreads();
  }
  if (threadIdx.x == 0) {
    float sin_t = sinf(theta_p[0]);
    float mean = red[0] / ((float)((size_t)O_DIM * I_DIM) * fabsf(sin_t));
    scale_out[0] = 1.0f / fmaxf(mean, 1e-5f);
  }
}

// ------- Pass 3a: one-time f32 -> f16 conversion + ternarization to ws -------
__global__ __launch_bounds__(256)
void convert_quant(const float* __restrict__ V, const float* __restrict__ Vp,
                   const float* __restrict__ X, const float* __restrict__ theta_p,
                   const float* __restrict__ scale_p,
                   _Float16* __restrict__ Xh, _Float16* __restrict__ Wph,
                   _Float16* __restrict__ Wth) {
  const float cos_t   = cosf(theta_p[0]);
  const float inv_sin = 1.0f / sinf(theta_p[0]);
  const float scale   = scale_p[0];
  const size_t nq = (size_t)O_DIM * I_DIM / 4;
  for (size_t q = (size_t)blockIdx.x * 256 + threadIdx.x; q < nq;
       q += (size_t)gridDim.x * 256) {
    size_t off = q * 4;
    float4 xv = *(const float4*)(X + off);
    v4h hx = { (_Float16)xv.x, (_Float16)xv.y, (_Float16)xv.z, (_Float16)xv.w };
    *(v4h*)(Xh + off) = hx;
    float4 vv = *(const float4*)(V + off);
    float4 pv = *(const float4*)(Vp + off);
    float q0 = fminf(fmaxf(rintf((vv.x - pv.x * cos_t) * inv_sin * scale), -1.f), 1.f);
    float q1 = fminf(fmaxf(rintf((vv.y - pv.y * cos_t) * inv_sin * scale), -1.f), 1.f);
    float q2 = fminf(fmaxf(rintf((vv.z - pv.z * cos_t) * inv_sin * scale), -1.f), 1.f);
    float q3 = fminf(fmaxf(rintf((vv.w - pv.w * cos_t) * inv_sin * scale), -1.f), 1.f);
    v4h hp = { (_Float16)pv.x, (_Float16)pv.y, (_Float16)pv.z, (_Float16)pv.w };
    v4h ht = { (_Float16)q0,   (_Float16)q1,   (_Float16)q2,   (_Float16)q3   };
    *(v4h*)(Wph + off) = hp;
    *(v4h*)(Wth + off) = ht;
  }
}

// ---- Pass 3b: dual GEMM on f16 operands, async-LDS double buffered, WMMA ----
__device__ __forceinline__ void issue_stage(unsigned int lds_base, int tid,
                                            const _Float16* Xh, const _Float16* Wph,
                                            const _Float16* Wth, int m0, int n0, int k0) {
  // X tile: 128 rows x 64 halves = 1024 x 16B chunks, 4 per thread
  #pragma unroll
  for (int i = 0; i < 4; i++) {
    int idx = tid + i * 256;
    int row = idx >> 3;
    int ch  = (idx & 7) * 8;  // column in halves (8 halves = 16B)
    unsigned int lda  = lds_base + (unsigned int)(row * LDSK2 + ch) * 2;
    unsigned int goff = (unsigned int)(((m0 + row) * I_DIM + k0 + ch) * 2);
    async_b128(lda, Xh, goff);
  }
  // Weight tiles: 64 rows x 64 halves each = 512 chunks, 2 per thread per matrix
  #pragma unroll
  for (int i = 0; i < 2; i++) {
    int idx = tid + i * 256;
    int row = idx >> 3;
    int ch  = (idx & 7) * 8;
    unsigned int loff = (unsigned int)(row * LDSK2 + ch) * 2;
    unsigned int goff = (unsigned int)(((n0 + row) * I_DIM + k0 + ch) * 2);
    async_b128(lds_base + WP_OFF * 2 + loff, Wph, goff);
    async_b128(lds_base + WT_OFF * 2 + loff, Wth, goff);
  }
}

__global__ __launch_bounds__(256)
void dual_gemm_wmma_async(const _Float16* __restrict__ Xh, const _Float16* __restrict__ Wph,
                          const _Float16* __restrict__ Wth, const float* __restrict__ theta_p,
                          const float* __restrict__ bias_p, float* __restrict__ out) {
  __shared__ __align__(16) _Float16 lds[2][STAGE_H];

  const float cos_t = cosf(theta_p[0]);
  const float bias  = bias_p[0];

  const int tid    = threadIdx.x;
  const int lane   = tid & 31;
  const int wave   = tid >> 5;
  const int wm     = wave & 3;
  const int wn     = wave >> 2;
  const int m_lane = lane & 15;
  const int kb     = (lane >> 4) * 8;

  const int m0 = blockIdx.y * BM;
  const int n0 = blockIdx.x * BN;

  const v8f vzero = {0.f, 0.f, 0.f, 0.f, 0.f, 0.f, 0.f, 0.f};
  v8f accT[2][2] = {{vzero, vzero}, {vzero, vzero}};
  v8f accP[2][2] = {{vzero, vzero}, {vzero, vzero}};

  const unsigned int lds_base0 = (unsigned int)(uintptr_t)&lds[0][0];
  const unsigned int lds_base1 = (unsigned int)(uintptr_t)&lds[1][0];

  // prologue: fill buffer 0
  issue_stage(lds_base0, tid, Xh, Wph, Wth, m0, n0, 0);
  wait_async0();
  __syncthreads();

  int buf = 0;
  for (int k0 = 0; k0 < I_DIM; k0 += BK2) {
    const int knext = k0 + BK2;
    if (knext < I_DIM)
      issue_stage(buf ? lds_base0 : lds_base1, tid, Xh, Wph, Wth, m0, n0, knext);

    const _Float16* L = &lds[buf][0];
    #pragma unroll
    for (int s = 0; s < 2; s++) {          // two 16x16x32 K-substeps per stage
      const int kc = s * 32 + kb;
      v16h afrag[2], bfp[2], bft[2];
      #pragma unroll
      for (int tm = 0; tm < 2; tm++) {
        int r = wm * 32 + tm * 16 + m_lane;
        v8h lo = *(const v8h*)&L[r * LDSK2 + kc];
        v8h hi = *(const v8h*)&L[r * LDSK2 + kc + 16];
        afrag[tm] = __builtin_shufflevector(lo, hi, 0,1,2,3,4,5,6,7,8,9,10,11,12,13,14,15);
      }
      #pragma unroll
      for (int tn = 0; tn < 2; tn++) {
        int r = wn * 32 + tn * 16 + m_lane;
        v8h lo = *(const v8h*)&L[WP_OFF + r * LDSK2 + kc];
        v8h hi = *(const v8h*)&L[WP_OFF + r * LDSK2 + kc + 16];
        bfp[tn] = __builtin_shufflevector(lo, hi, 0,1,2,3,4,5,6,7,8,9,10,11,12,13,14,15);
        lo = *(const v8h*)&L[WT_OFF + r * LDSK2 + kc];
        hi = *(const v8h*)&L[WT_OFF + r * LDSK2 + kc + 16];
        bft[tn] = __builtin_shufflevector(lo, hi, 0,1,2,3,4,5,6,7,8,9,10,11,12,13,14,15);
      }
      #pragma unroll
      for (int tm = 0; tm < 2; tm++) {
        #pragma unroll
        for (int tn = 0; tn < 2; tn++) {
          accP[tm][tn] = __builtin_amdgcn_wmma_f32_16x16x32_f16(
              false, afrag[tm], false, bfp[tn], (short)0, accP[tm][tn], false, false);
          accT[tm][tn] = __builtin_amdgcn_wmma_f32_16x16x32_f16(
              false, afrag[tm], false, bft[tn], (short)0, accT[tm][tn], false, false);
        }
      }
    }
    wait_async0();     // my copies for the next stage are complete
    __syncthreads();   // everyone's copies complete / reads of old buffer done
    buf ^= 1;
  }

  const int n_lane = lane & 15;
  const int m_half = (lane >> 4) * 8;
  #pragma unroll
  for (int tm = 0; tm < 2; tm++) {
    #pragma unroll
    for (int tn = 0; tn < 2; tn++) {
      int gm_base = m0 + wm * 32 + tm * 16 + m_half;
      int gn      = n0 + wn * 32 + tn * 16 + n_lane;
      #pragma unroll
      for (int r = 0; r < 8; r++) {
        float t = accT[tm][tn][r];
        float s = (t > 0.f) ? 1.f : ((t < 0.f) ? -1.f : 0.f);
        out[(size_t)(gm_base + r) * O_DIM + gn] = cos_t * accP[tm][tn][r] + s * bias;
      }
    }
  }
}

// -------- Fallback fused dual GEMM (on-the-fly f32->f16, small ws) -----------
__global__ __launch_bounds__(256)
void dual_gemm_wmma(const float* __restrict__ V, const float* __restrict__ Vp,
                    const float* __restrict__ X, const float* __restrict__ theta_p,
                    const float* __restrict__ bias_p, const float* __restrict__ scale_p,
                    float* __restrict__ out) {
  __shared__ __align__(16) _Float16 Xs[BM][LDSK];
  __shared__ __align__(16) _Float16 Wp[BN][LDSK];
  __shared__ __align__(16) _Float16 Wt[BN][LDSK];

  const float theta   = theta_p[0];
  const float cos_t   = cosf(theta);
  const float inv_sin = 1.0f / sinf(theta);
  const float scale   = scale_p[0];
  const float bias    = bias_p[0];

  const int tid    = threadIdx.x;
  const int lane   = tid & 31;
  const int wave   = tid >> 5;
  const int wm     = wave & 3;
  const int wn     = wave >> 2;
  const int m_lane = lane & 15;
  const int kb     = (lane >> 4) * 8;

  const int m0 = blockIdx.y * BM;
  const int n0 = blockIdx.x * BN;

  const v8f vzero = {0.f, 0.f, 0.f, 0.f, 0.f, 0.f, 0.f, 0.f};
  v8f accT[2][2] = {{vzero, vzero}, {vzero, vzero}};
  v8f accP[2][2] = {{vzero, vzero}, {vzero, vzero}};

  for (int k0 = 0; k0 < I_DIM; k0 += BK) {
    #pragma unroll
    for (int i = 0; i < 4; i++) {
      int idx = tid + i * 256;
      int row = idx >> 3;
      int c4  = (idx & 7) * 4;
      float4 xv = *(const float4*)(X + (size_t)(m0 + row) * I_DIM + k0 + c4);
      v4h h = { (_Float16)xv.x, (_Float16)xv.y, (_Float16)xv.z, (_Float16)xv.w };
      *(v4h*)&Xs[row][c4] = h;
    }
    #pragma unroll
    for (int i = 0; i < 2; i++) {
      int idx = tid + i * 256;
      int row = idx >> 3;
      int c4  = (idx & 7) * 4;
      size_t goff = (size_t)(n0 + row) * I_DIM + k0 + c4;
      float4 vv = *(const float4*)(V + goff);
      float4 pv = *(const float4*)(Vp + goff);
      float q0 = fminf(fmaxf(rintf((vv.x - pv.x * cos_t) * inv_sin * scale), -1.f), 1.f);
      float q1 = fminf(fmaxf(rintf((vv.y - pv.y * cos_t) * inv_sin * scale), -1.f), 1.f);
      float q2 = fminf(fmaxf(rintf((vv.z - pv.z * cos_t) * inv_sin * scale), -1.f), 1.f);
      float q3 = fminf(fmaxf(rintf((vv.w - pv.w * cos_t) * inv_sin * scale), -1.f), 1.f);
      v4h hp = { (_Float16)pv.x, (_Float16)pv.y, (_Float16)pv.z, (_Float16)pv.w };
      v4h ht = { (_Float16)q0,   (_Float16)q1,   (_Float16)q2,   (_Float16)q3   };
      *(v4h*)&Wp[row][c4] = hp;
      *(v4h*)&Wt[row][c4] = ht;
    }
    __syncthreads();

    v16h afrag[2], bfp[2], bft[2];
    #pragma unroll
    for (int tm = 0; tm < 2; tm++) {
      int r = wm * 32 + tm * 16 + m_lane;
      v8h lo = *(const v8h*)&Xs[r][kb];
      v8h hi = *(const v8h*)&Xs[r][kb + 16];
      afrag[tm] = __builtin_shufflevector(lo, hi, 0,1,2,3,4,5,6,7,8,9,10,11,12,13,14,15);
    }
    #pragma unroll
    for (int tn = 0; tn < 2; tn++) {
      int r = wn * 32 + tn * 16 + m_lane;
      v8h lo = *(const v8h*)&Wp[r][kb];
      v8h hi = *(const v8h*)&Wp[r][kb + 16];
      bfp[tn] = __builtin_shufflevector(lo, hi, 0,1,2,3,4,5,6,7,8,9,10,11,12,13,14,15);
      lo = *(const v8h*)&Wt[r][kb];
      hi = *(const v8h*)&Wt[r][kb + 16];
      bft[tn] = __builtin_shufflevector(lo, hi, 0,1,2,3,4,5,6,7,8,9,10,11,12,13,14,15);
    }
    #pragma unroll
    for (int tm = 0; tm < 2; tm++) {
      #pragma unroll
      for (int tn = 0; tn < 2; tn++) {
        accP[tm][tn] = __builtin_amdgcn_wmma_f32_16x16x32_f16(
            false, afrag[tm], false, bfp[tn], (short)0, accP[tm][tn], false, false);
        accT[tm][tn] = __builtin_amdgcn_wmma_f32_16x16x32_f16(
            false, afrag[tm], false, bft[tn], (short)0, accT[tm][tn], false, false);
      }
    }
    __syncthreads();
  }

  const int n_lane = lane & 15;
  const int m_half = (lane >> 4) * 8;
  #pragma unroll
  for (int tm = 0; tm < 2; tm++) {
    #pragma unroll
    for (int tn = 0; tn < 2; tn++) {
      int gm_base = m0 + wm * 32 + tm * 16 + m_half;
      int gn      = n0 + wn * 32 + tn * 16 + n_lane;
      #pragma unroll
      for (int r = 0; r < 8; r++) {
        float t = accT[tm][tn][r];
        float s = (t > 0.f) ? 1.f : ((t < 0.f) ? -1.f : 0.f);
        out[(size_t)(gm_base + r) * O_DIM + gn] = cos_t * accP[tm][tn][r] + s * bias;
      }
    }
  }
}

extern "C" void kernel_launch(void* const* d_in, const int* in_sizes, int n_in,
                              void* d_out, int out_size, void* d_ws, size_t ws_size,
                              hipStream_t stream) {
  (void)in_sizes; (void)n_in; (void)out_size;
  const float* V     = (const float*)d_in[0];
  const float* Vp    = (const float*)d_in[1];
  const float* X     = (const float*)d_in[2];
  const float* theta = (const float*)d_in[3];
  const float* bias  = (const float*)d_in[4];
  float* out = (float*)d_out;

  float* scale    = (float*)d_ws;                        // 1 float
  float* partials = (float*)((char*)d_ws + 256);         // RED_BLOCKS floats

  const size_t mat_h = (size_t)O_DIM * I_DIM;            // elements per matrix
  const size_t f16_base = 16384;
  const size_t ws_need  = f16_base + 3 * mat_h * sizeof(_Float16);

  absmean_partial<<<RED_BLOCKS, RED_THREADS, 0, stream>>>(V, Vp, theta, partials);
  finalize_scale<<<1, RED_THREADS, 0, stream>>>(partials, theta, scale);

  dim3 grid(O_DIM / BN, B_DIM / BM);
  if (ws_size >= ws_need) {
    _Float16* Xh  = (_Float16*)((char*)d_ws + f16_base);
    _Float16* Wph = Xh + mat_h;
    _Float16* Wth = Wph + mat_h;
    convert_quant<<<4096, 256, 0, stream>>>(V, Vp, X, theta, scale, Xh, Wph, Wth);
    dual_gemm_wmma_async<<<grid, 256, 0, stream>>>(Xh, Wph, Wth, theta, bias, out);
  } else {
    dual_gemm_wmma<<<grid, 256, 0, stream>>>(V, Vp, X, theta, bias, scale, out);
  }
}
